// Policy_31894427140491
// MI455X (gfx1250) — matrix operand order repaired and checked by reading, not compiled, not run on hardware
//
#include <hip/hip_runtime.h>
#include <hip/hip_bf16.h>
#include <math.h>

// Problem constants (from reference): T=256, N=128, H=512, D=512, A=2
#define T_STEPS 256
#define NENV    128
#define HDIM    512
#define TN      (T_STEPS * NENV)      // 32768
#define G3H     (3 * HDIM)            // 1536

#define USE_TDM 1                     // Tensor Data Mover staging of B tiles

typedef unsigned short u16;
typedef __attribute__((ext_vector_type(16))) __bf16 v16bf;
typedef __attribute__((ext_vector_type(8)))  float  v8f;
typedef __attribute__((ext_vector_type(4)))  unsigned v4u;
typedef __attribute__((ext_vector_type(8)))  unsigned v8u;

union Frag { v16bf v; uint4 q[2]; };

__device__ __forceinline__ u16 f2bf(float x) {
  union { float f; unsigned u; } c; c.f = x;
  unsigned u = c.u + 0x7fffu + ((c.u >> 16) & 1u);   // round-to-nearest-even
  return (u16)(u >> 16);
}
__device__ __forceinline__ float bf2f(u16 h) {
  union { unsigned u; float f; } c; c.u = ((unsigned)h) << 16;
  return c.f;
}
__device__ __forceinline__ float sigmoidf(float x) { return 1.0f / (1.0f + expf(-x)); }

__device__ __forceinline__ v8f wmma_bf16(Frag a, Frag b, v8f c) {
  // D = A(16x32 bf16) * B(32x16 bf16) + C(f32)
  return __builtin_amdgcn_wmma_f32_16x16x32_bf16(false, a.v, false, b.v, (short)0, c, false, false);
}

#if USE_TDM
// Issue one 2-D TDM load: `rows` rows of `rowHalves` bf16 elements from a
// row-major matrix with row stride `strideHalves`, into LDS at byte offset
// `ldsOff`, inserting pad so the LDS row pitch = data + pad (bank-conflict
// avoidance).  padIntervalCode / padAmountCode per CDNA5 D# group1 encoding:
//   interval: 0=2,1=4,2=8,3=16,4=32,5=64,...,7=256 DWORDs before each pad
//   amount:   code+1 DWORDs of pad
__device__ __forceinline__ void tdm_load_2d(unsigned ldsOff, const u16* gptr,
                                            unsigned rows, unsigned rowHalves,
                                            unsigned strideHalves,
                                            unsigned padIntervalCode,
                                            unsigned padAmountCode) {
  unsigned long long ga = (unsigned long long)(size_t)gptr;
  v4u g0;
  g0.x = 1u;                                  // count=1, user descriptor
  g0.y = ldsOff;                              // lds_addr [63:32]
  g0.z = (unsigned)ga;                        // global_addr low
  g0.w = ((unsigned)(ga >> 32) & 0x01FFFFFFu) | (2u << 30);  // addr hi | type=2
  v8u g1;
  g1[0] = (1u << 16)                          // data_size = 2 bytes
        | (1u << 20)                          // pad_enable
        | (padIntervalCode << 22)
        | (padAmountCode << 25);
  g1[1] = (rowHalves << 16);                  // tensor_dim0 (bits 79:48)
  g1[2] = (rows << 16);                       // tensor_dim1 (bits 111:80)
  g1[3] = (rowHalves << 16);                  // tile_dim0   (bits 127:112)
  g1[4] = rows;                               // tile_dim1 ; tile_dim2 = 0
  g1[5] = strideHalves;                       // tensor_dim0_stride (bits 207:160)
  g1[6] = 0u;
  g1[7] = 0u;                                 // tensor_dim1_stride unused (2D)
  asm volatile("tensor_load_to_lds %0, %1" :: "s"(g0), "s"(g1) : "memory");
}
#endif

// ---------------------------------------------------------------------------
// Elementwise f32 -> bf16 conversion (vectorized x4; all sizes divisible by 4)
// ---------------------------------------------------------------------------
__global__ __launch_bounds__(256) void cvt_f32_bf16(const float* __restrict__ s,
                                                    u16* __restrict__ d, int n4) {
  int i = blockIdx.x * 256 + threadIdx.x;
  if (i < n4) {
    float4 f = ((const float4*)s)[i];
    ((ushort4*)d)[i] = make_ushort4(f2bf(f.x), f2bf(f.y), f2bf(f.z), f2bf(f.w));
  }
}

// h0 -> fp32 state buffer 0 and bf16 state buffer 0
__global__ __launch_bounds__(256) void init_h(const float* __restrict__ h0,
                                              float* __restrict__ hf,
                                              u16* __restrict__ hb, int n) {
  int i = blockIdx.x * 256 + threadIdx.x;
  if (i < n) { float v = h0[i]; hf[i] = v; hb[i] = f2bf(v); }
}

// ---------------------------------------------------------------------------
// WMMA GEMM:  C[M,N] = act( A[M,K](bf16) @ W[N,K]^T(bf16) + bias[N] ), C bf16
// Workgroup: 256 threads = 8 waves; tile = 128 rows x 64 cols; K step 32.
// B tile staged in LDS via TDM (64 rows x 32 halves, pitch 40 halves = 80B:
// 64B data + 16B pad -> 16 distinct banks across the 16 fragment rows).
// ---------------------------------------------------------------------------
template <int ACT>   // 0 = none, 1 = tanh
__global__ __launch_bounds__(256) void gemm_bf16(const u16* __restrict__ A,
                                                 const u16* __restrict__ W,
                                                 const float* __restrict__ bias,
                                                 u16* __restrict__ C,
                                                 int M, int N, int K) {
  __shared__ __align__(16) u16 ldsB[64 * 40];
  const int tid  = threadIdx.x;
  const int wave = tid >> 5;
  const int lane = tid & 31;
  const int l16  = lane & 15;
  const int lh   = lane >> 4;                 // 0 or 1
  const int mbase = blockIdx.x * 128 + wave * 16;
  const int jb    = blockIdx.y * 64;

  v8f acc[4] = {};
  const u16* aRow = A + (size_t)(mbase + l16) * K + lh * 8;

  for (int k0 = 0; k0 < K; k0 += 32) {
#if USE_TDM
    if (wave == 0) {
      // 64 rows x 64B: pad_interval code 3 (16 DW), pad_amount code 3 (4 DW)
      tdm_load_2d((unsigned)(size_t)&ldsB[0], W + (size_t)jb * K + k0,
                  64u, 32u, (unsigned)K, 3u, 3u);
      __builtin_amdgcn_s_wait_tensorcnt(0);
    }
#else
    { // cooperative B-tile load: 64 rows x 4 x 16B segments = 256 loads
      int r = tid >> 2, s = tid & 3;
      uint4 src = *(const uint4*)(W + (size_t)(jb + r) * K + k0 + s * 8);
      *(uint4*)(&ldsB[r * 40 + s * 8]) = src;
    }
#endif
    if (k0 + 32 < K) __builtin_prefetch(aRow + k0 + 32, 0, 1);
    __syncthreads();

    Frag a;
    a.q[0] = *(const uint4*)(aRow + k0);        // K = k0 + lh*8 + 0..7
    a.q[1] = *(const uint4*)(aRow + k0 + 16);   // K = k0 + 16 + lh*8 + 0..7
#pragma unroll
    for (int c = 0; c < 4; ++c) {
      Frag b;
      const u16* bp = &ldsB[(c * 16 + l16) * 40 + lh * 8];
      b.q[0] = *(const uint4*)(bp);
      b.q[1] = *(const uint4*)(bp + 16);
      acc[c] = wmma_bf16(a, b, acc[c]);
    }
    __syncthreads();
  }

#pragma unroll
  for (int c = 0; c < 4; ++c) {
    int col = jb + c * 16 + l16;
    float bv = bias[col];
#pragma unroll
    for (int r = 0; r < 8; ++r) {
      int row = mbase + lh * 8 + r;
      float v = acc[c][r] + bv;
      if (ACT == 1) v = tanhf(v);
      C[(size_t)row * N + col] = f2bf(v);
    }
  }
}

// ---------------------------------------------------------------------------
// One GRU time step for BOTH GRUs (blockIdx.y: 0=actor, 1=critic).
// blockIdx.x selects a 64-wide hidden-unit column block (8 blocks cover H).
// gh tiles for gates r,z,n ([128 x 64] each, K=512) via WMMA; Whh staged in
// K-chunks of 128 halves (192 rows x 336B pitch = 63KB LDS, 4 chunks/step),
// then fused gate math with double-buffered h (fp32 + bf16).
// Actor writes ys rows into d_out; critic emits deterministic value partials.
// ---------------------------------------------------------------------------
#define GPITCH 168   // LDS row pitch in halves (336B = 256B data + 80B pad)

__global__ __launch_bounds__(256) void gru_step(
    const u16* __restrict__ gxa, const u16* __restrict__ gxc,
    const u16* __restrict__ whhA, const u16* __restrict__ whhC,
    const float* __restrict__ bhhA, const float* __restrict__ bhhC,
    const float* __restrict__ masks,
    float* __restrict__ hfA, float* __restrict__ hfC,
    u16* __restrict__ hbA, u16* __restrict__ hbC,
    float* __restrict__ hid_actor,      // d_out + 32768, [TN x H], row = n*T + t
    const float* __restrict__ Wv, float* __restrict__ vpart,
    int t) {
  const bool actor = (blockIdx.y == 0);
  const u16*   gx  = actor ? gxa  : gxc;
  const u16*   whh = actor ? whhA : whhC;
  const float* bhh = actor ? bhhA : bhhC;
  float* hf = actor ? hfA : hfC;
  u16*   hb = actor ? hbA : hbC;

  const int rd = t & 1;                 // read buffer
  const int wr = rd ^ 1;                // write buffer
  const float* hf_rd = hf + (size_t)rd * NENV * HDIM;
  float*       hf_wr = hf + (size_t)wr * NENV * HDIM;
  const u16*   hb_rd = hb + (size_t)rd * NENV * HDIM;
  u16*         hb_wr = hb + (size_t)wr * NENV * HDIM;

  __shared__ __align__(16) u16 ldsB[192 * GPITCH];   // 64512 bytes
  const int tid  = threadIdx.x;
  const int wave = tid >> 5;
  const int lane = tid & 31;
  const int l16  = lane & 15;
  const int lh   = lane >> 4;
  const int jb   = blockIdx.x * 64;

  // ---- gh = (mask * h_prev) @ Whh^T  (three 128x64 tiles) ----
  const int nrowA = wave * 16 + l16;                    // env row for A frag
  const float mA  = masks[t * NENV + nrowA];            // 0.0 or 1.0
  const u16* aRow = hb_rd + (size_t)nrowA * HDIM + lh * 8;

  v8f acc[3][4] = {};
  for (int kc = 0; kc < HDIM; kc += 128) {
#if USE_TDM
    if (wave == 0) {
      // per gate: 64 rows x 256B: interval code 5 (64 DW), amount code 19 (20 DW)
#pragma unroll
      for (int g = 0; g < 3; ++g) {
        tdm_load_2d((unsigned)(size_t)&ldsB[g * 64 * GPITCH],
                    whh + (size_t)(g * HDIM + jb) * HDIM + kc,
                    64u, 128u, (unsigned)HDIM, 5u, 19u);
      }
      __builtin_amdgcn_s_wait_tensorcnt(0);
    }
#else
    // cooperative: 192 rows x 16 x 16B segments = 3072 loads / 256 thr = 12
#pragma unroll
    for (int it = 0; it < 12; ++it) {
      int idx = tid + it * 256;
      int r = idx >> 4, s = idx & 15;    // row 0..191, 16B seg 0..15
      int g = r >> 6, rr = r & 63;
      uint4 src = *(const uint4*)(whh + (size_t)(g * HDIM + jb + rr) * HDIM + kc + s * 8);
      *(uint4*)(&ldsB[r * GPITCH + s * 8]) = src;
    }
#endif
    __syncthreads();

#pragma unroll
    for (int k0 = 0; k0 < 128; k0 += 32) {
      Frag a;
      if (mA != 0.0f) {
        a.q[0] = *(const uint4*)(aRow + kc + k0);
        a.q[1] = *(const uint4*)(aRow + kc + k0 + 16);
      } else {
        a.q[0] = make_uint4(0, 0, 0, 0);
        a.q[1] = make_uint4(0, 0, 0, 0);
      }
#pragma unroll
      for (int g = 0; g < 3; ++g)
#pragma unroll
        for (int c = 0; c < 4; ++c) {
          Frag b;
          const u16* bp = &ldsB[(g * 64 + c * 16 + l16) * GPITCH + k0 + lh * 8];
          b.q[0] = *(const uint4*)(bp);
          b.q[1] = *(const uint4*)(bp + 16);
          acc[g][c] = wmma_bf16(a, b, acc[g][c]);
        }
    }
    __syncthreads();
  }

  // ---- fused gate math + state update ----
#pragma unroll
  for (int r = 0; r < 8; ++r) {
    const int n    = wave * 16 + lh * 8 + r;       // env index of this C row
    const float mn = masks[t * NENV + n];
    const int mrow = n * T_STEPS + t;              // row in gx / outputs
    float vsum = 0.0f;
#pragma unroll
    for (int c = 0; c < 4; ++c) {
      const int j = jb + c * 16 + l16;             // hidden unit
      float ghr = acc[0][c][r] + bhh[j];
      float ghz = acc[1][c][r] + bhh[HDIM + j];
      float ghn = acc[2][c][r] + bhh[2 * HDIM + j];
      const size_t gb = (size_t)mrow * G3H;
      float gxr = bf2f(gx[gb + j]);
      float gxz = bf2f(gx[gb + HDIM + j]);
      float gxn = bf2f(gx[gb + 2 * HDIM + j]);
      float rg = sigmoidf(gxr + ghr);
      float zg = sigmoidf(gxz + ghz);
      float ng = tanhf(gxn + rg * ghn);
      float hm = (mn != 0.0f) ? hf_rd[(size_t)n * HDIM + j] : 0.0f;
      float hn = (1.0f - zg) * ng + zg * hm;
      hf_wr[(size_t)n * HDIM + j] = hn;
      hb_wr[(size_t)n * HDIM + j] = f2bf(hn);
      if (actor) {
        hid_actor[(size_t)mrow * HDIM + j] = hn;   // ys_a
      } else {
        vsum += hn * Wv[j];                        // partial value dot
      }
    }
    if (!actor) {
      // reduce over the 16 column lanes (wave32; 16-lane halves)
      for (int off = 1; off < 16; off <<= 1) vsum += __shfl_xor(vsum, off, 32);
      if (l16 == 0) vpart[(size_t)mrow * 8 + blockIdx.x] = vsum;
    }
  }
}

// value[m] = sum_blk vpart[m][blk] + bv   (deterministic reduction)
__global__ __launch_bounds__(256) void value_reduce(const float* __restrict__ vpart,
                                                    const float* __restrict__ bv,
                                                    float* __restrict__ value) {
  int m = blockIdx.x * 256 + threadIdx.x;
  if (m < TN) {
    float s = bv[0];
#pragma unroll
    for (int i = 0; i < 8; ++i) s += vpart[(size_t)m * 8 + i];
    value[m] = s;
  }
}

// ---------------------------------------------------------------------------
extern "C" void kernel_launch(void* const* d_in, const int* in_sizes, int n_in,
                              void* d_out, int out_size, void* d_ws, size_t ws_size,
                              hipStream_t stream) {
  (void)in_sizes; (void)n_in; (void)out_size; (void)ws_size;

  const float* share  = (const float*)d_in[0];   // [TN, 1024]
  const float* inputs = (const float*)d_in[1];   // [TN, 512]
  const float* h0a    = (const float*)d_in[2];
  const float* h0c    = (const float*)d_in[3];
  const float* masks  = (const float*)d_in[4];
  const float* Wa1 = (const float*)d_in[5],  *ba1 = (const float*)d_in[6];
  const float* Wa2 = (const float*)d_in[7],  *ba2 = (const float*)d_in[8];
  const float* Wc1 = (const float*)d_in[9],  *bc1 = (const float*)d_in[10];
  const float* Wc2 = (const float*)d_in[11], *bc2 = (const float*)d_in[12];
  const float* Wih_a = (const float*)d_in[13], *Whh_a = (const float*)d_in[14];
  const float* bih_a = (const float*)d_in[15], *bhh_a = (const float*)d_in[16];
  const float* Wih_c = (const float*)d_in[17], *Whh_c = (const float*)d_in[18];
  const float* bih_c = (const float*)d_in[19], *bhh_c = (const float*)d_in[20];
  const float* Wv = (const float*)d_in[21], *bv = (const float*)d_in[22];

  // ---- workspace carve-up ----
  char* base = (char*)d_ws;
  size_t off = 0;
  auto alloc = [&](size_t bytes) -> void* {
    void* p = base + off;
    off = (off + bytes + 255) & ~(size_t)255;
    return p;
  };
  u16* bWa1  = (u16*)alloc((size_t)512 * 512 * 2);
  u16* bWa2  = (u16*)alloc((size_t)512 * 512 * 2);
  u16* bWc1  = (u16*)alloc((size_t)512 * 1024 * 2);
  u16* bWc2  = (u16*)alloc((size_t)512 * 512 * 2);
  u16* bWihA = (u16*)alloc((size_t)G3H * 512 * 2);
  u16* bWhhA = (u16*)alloc((size_t)G3H * 512 * 2);
  u16* bWihC = (u16*)alloc((size_t)G3H * 512 * 2);
  u16* bWhhC = (u16*)alloc((size_t)G3H * 512 * 2);
  u16* bXa   = (u16*)alloc((size_t)TN * 512 * 2);
  u16* bXc   = (u16*)alloc((size_t)TN * 1024 * 2);
  u16* bT1   = (u16*)alloc((size_t)TN * 512 * 2);    // layer-1 output (reused)
  u16* bHA   = (u16*)alloc((size_t)TN * 512 * 2);
  u16* bHC   = (u16*)alloc((size_t)TN * 512 * 2);
  u16* bGXa  = (u16*)alloc((size_t)TN * G3H * 2);
  u16* bGXc  = (u16*)alloc((size_t)TN * G3H * 2);
  float* hfA = (float*)alloc((size_t)2 * NENV * HDIM * 4);  // double buffer
  float* hfC = (float*)alloc((size_t)2 * NENV * HDIM * 4);
  u16*   hbA = (u16*)alloc((size_t)2 * NENV * HDIM * 2);
  u16*   hbC = (u16*)alloc((size_t)2 * NENV * HDIM * 2);
  float* vpart = (float*)alloc((size_t)TN * 8 * 4);

  auto cvt = [&](const float* s, u16* d, size_t n) {
    int n4 = (int)(n / 4);
    cvt_f32_bf16<<<dim3((n4 + 255) / 256), dim3(256), 0, stream>>>(s, d, n4);
  };

  // 1) precision conversion
  cvt(inputs, bXa, (size_t)TN * 512);
  cvt(share,  bXc, (size_t)TN * 1024);
  cvt(Wa1, bWa1, (size_t)512 * 512);
  cvt(Wa2, bWa2, (size_t)512 * 512);
  cvt(Wc1, bWc1, (size_t)512 * 1024);
  cvt(Wc2, bWc2, (size_t)512 * 512);
  cvt(Wih_a, bWihA, (size_t)G3H * 512);
  cvt(Whh_a, bWhhA, (size_t)G3H * 512);
  cvt(Wih_c, bWihC, (size_t)G3H * 512);
  cvt(Whh_c, bWhhC, (size_t)G3H * 512);
  init_h<<<dim3((NENV * HDIM + 255) / 256), dim3(256), 0, stream>>>(h0a, hfA, hbA, NENV * HDIM);
  init_h<<<dim3((NENV * HDIM + 255) / 256), dim3(256), 0, stream>>>(h0c, hfC, hbC, NENV * HDIM);

  // 2) MLPs (tanh) and GRU input-gate precompute (no act)
  gemm_bf16<1><<<dim3(TN / 128, 512 / 64), dim3(256), 0, stream>>>(bXa, bWa1, ba1, bT1, TN, 512, 512);
  gemm_bf16<1><<<dim3(TN / 128, 512 / 64), dim3(256), 0, stream>>>(bT1, bWa2, ba2, bHA, TN, 512, 512);
  gemm_bf16<1><<<dim3(TN / 128, 512 / 64), dim3(256), 0, stream>>>(bXc, bWc1, bc1, bT1, TN, 512, 1024);
  gemm_bf16<1><<<dim3(TN / 128, 512 / 64), dim3(256), 0, stream>>>(bT1, bWc2, bc2, bHC, TN, 512, 512);
  gemm_bf16<0><<<dim3(TN / 128, G3H / 64), dim3(256), 0, stream>>>(bHA, bWihA, bih_a, bGXa, TN, G3H, 512);
  gemm_bf16<0><<<dim3(TN / 128, G3H / 64), dim3(256), 0, stream>>>(bHC, bWihC, bih_c, bGXc, TN, G3H, 512);

  // 3) sequential GRU scan (both GRUs per launch; double-buffered state)
  float* out = (float*)d_out;
  float* hid_actor = out + TN;                           // [TN x H]
  for (int t = 0; t < T_STEPS; ++t) {
    gru_step<<<dim3(HDIM / 64, 2), dim3(256), 0, stream>>>(
        bGXa, bGXc, bWhhA, bWhhC, bhh_a, bhh_c, masks,
        hfA, hfC, hbA, hbC, hid_actor, Wv, vpart, t);
  }

  // 4) value head + final hidden states (final h lands in buffer 0)
  value_reduce<<<dim3(TN / 256), dim3(256), 0, stream>>>(vpart, bv, out);
  float* hTa_dst = out + TN + (size_t)TN * HDIM;
  float* hTc_dst = hTa_dst + NENV * HDIM;
  hipMemcpyAsync(hTa_dst, hfA, (size_t)NENV * HDIM * 4, hipMemcpyDeviceToDevice, stream);
  hipMemcpyAsync(hTc_dst, hfC, (size_t)NENV * HDIM * 4, hipMemcpyDeviceToDevice, stream);
}
